// QuantizedLinearWithPara_87522843559263
// MI455X (gfx1250) — matrix operand order repaired and checked by reading, not compiled, not run on hardware
//
#include <hip/hip_runtime.h>
#include <stdint.h>

// CDNA5 (gfx1250) wave32 WMMA types
typedef __attribute__((ext_vector_type(16))) __bf16 v16bf;
typedef __attribute__((ext_vector_type(8)))  float  v8f;

#define BM 128
#define BN 128
#define BK 32
#define BKP 40            // padded row length: 80B = 20 dwords -> conflict-free row stride
#define NTHREADS 256

union FragU {
    v16bf v;
    uint4 q[2];
};
union PackU {
    __bf16 b[16];
    uint4  q[2];
};

// C[M,N] = X[M,K] (fp32, split to bf16 hi+lo) * W[K,N] (int8 exact in bf16)
__global__ __launch_bounds__(NTHREADS)
void qlinear_wmma_bf16x2(const float* __restrict__ X,
                         const int8_t* __restrict__ W,
                         float* __restrict__ C,
                         int M, int N, int K)
{
    // Double-buffered LDS tiles, rows padded to 40 elems (20-dword stride, gcd(20,64)=4
    // -> 16 consecutive rows map to 16 distinct bank offsets, b128-aligned row bases).
    __shared__ alignas(16) __bf16 sAhi[2][BM][BKP];   // [m][k]
    __shared__ alignas(16) __bf16 sAlo[2][BM][BKP];   // [m][k]
    __shared__ alignas(16) __bf16 sB  [2][BN][BKP];   // [n][k] (transposed)

    const int tid  = threadIdx.x;
    const int lane = tid & 31;
    const int wave = tid >> 5;

    const int bm = blockIdx.y * BM;
    const int bn = blockIdx.x * BN;

    // A staging: thread t loads 16 consecutive fp32 of one row (64B)
    const int arow = tid >> 1;          // 0..127
    const int acol = (tid & 1) * 16;    // 0 or 16
    // B staging: thread t loads 16 consecutive int8 of one K-row (16B)
    const int brow = tid >> 3;          // 0..31 (k)
    const int bgrp = tid & 7;           // n-group: n = bgrp*16 + e
    const int brot = ((brow & 3) << 2) + bgrp;  // per-thread store-order rotation

    // Wave tile: 2 (M) x 4 (N) waves; each wave 64x32 of output
    const int wm = (wave >> 2) * 64;    // 0 or 64
    const int wn = (wave & 3)  * 32;    // 0,32,64,96

    v8f acc[4][2];
#pragma unroll
    for (int mt = 0; mt < 4; ++mt)
#pragma unroll
        for (int nt = 0; nt < 2; ++nt)
            acc[mt][nt] = (v8f)(0.0f);

    // ---- staging helper: global -> (convert) -> LDS buffer `buf` ----
    auto stage = [&](int kbase, int buf) {
        // X: 16 floats -> bf16 hi + bf16 residual lo
        const float4* xp =
            reinterpret_cast<const float4*>(X + (size_t)(bm + arow) * K + kbase + acol);
        float4 xr0 = xp[0], xr1 = xp[1], xr2 = xp[2], xr3 = xp[3];
        float xf[16] = { xr0.x, xr0.y, xr0.z, xr0.w,
                         xr1.x, xr1.y, xr1.z, xr1.w,
                         xr2.x, xr2.y, xr2.z, xr2.w,
                         xr3.x, xr3.y, xr3.z, xr3.w };
        PackU hp, lp;
#pragma unroll
        for (int j = 0; j < 16; ++j) {
            float  f = xf[j];
            __bf16 h = (__bf16)f;
            hp.b[j] = h;
            lp.b[j] = (__bf16)(f - (float)h);
        }
        *reinterpret_cast<uint4*>(&sAhi[buf][arow][acol])     = hp.q[0];
        *reinterpret_cast<uint4*>(&sAhi[buf][arow][acol + 8]) = hp.q[1];
        *reinterpret_cast<uint4*>(&sAlo[buf][arow][acol])     = lp.q[0];
        *reinterpret_cast<uint4*>(&sAlo[buf][arow][acol + 8]) = lp.q[1];

        // W: 16 int8 from row-major [k][n] -> bf16 transposed LDS [n][k].
        // Store order rotated per-thread so concurrent lanes hit distinct banks.
        const int4 wv =
            *reinterpret_cast<const int4*>(W + (size_t)(kbase + brow) * N + bn + bgrp * 16);
        const int8_t* wc = reinterpret_cast<const int8_t*>(&wv);
#pragma unroll
        for (int j = 0; j < 16; ++j) {
            const int e = (j + brot) & 15;
            sB[buf][bgrp * 16 + e][brow] = (__bf16)(float)(int)wc[e];
        }
    };

    stage(0, 0);
    __syncthreads();

    const int ksteps = K / BK;
    for (int ks = 0; ks < ksteps; ++ks) {
        const int buf = ks & 1;
        if (ks + 1 < ksteps)
            stage((ks + 1) * BK, buf ^ 1);

        // B fragments (32x16 bf16): lane holds col n=lane%16,
        // K 0..15 (lanes 0-15) or 16..31 (lanes 16-31) -> 32B contiguous in sB[n][k]
        FragU bfr[2];
#pragma unroll
        for (int nt = 0; nt < 2; ++nt) {
            const int coln = wn + nt * 16 + (lane & 15);
            const int k0   = (lane & 16) ? 16 : 0;
            bfr[nt].q[0] = *reinterpret_cast<const uint4*>(&sB[buf][coln][k0]);
            bfr[nt].q[1] = *reinterpret_cast<const uint4*>(&sB[buf][coln][k0 + 8]);
        }

#pragma unroll
        for (int mt = 0; mt < 4; ++mt) {
            // A fragments (16x32 bf16): lane holds row m=lane%16,
            // K 0..15 (lanes 0-15) or 16..31 (lanes 16-31)
            const int row = wm + mt * 16 + (lane & 15);
            const int k0  = (lane & 16) ? 16 : 0;
            FragU ahi, alo;
            ahi.q[0] = *reinterpret_cast<const uint4*>(&sAhi[buf][row][k0]);
            ahi.q[1] = *reinterpret_cast<const uint4*>(&sAhi[buf][row][k0 + 8]);
            alo.q[0] = *reinterpret_cast<const uint4*>(&sAlo[buf][row][k0]);
            alo.q[1] = *reinterpret_cast<const uint4*>(&sAlo[buf][row][k0 + 8]);
#pragma unroll
            for (int nt = 0; nt < 2; ++nt) {
                acc[mt][nt] = __builtin_amdgcn_wmma_f32_16x16x32_bf16(
                    false, ahi.v, false, bfr[nt].v, (short)0, acc[mt][nt], false, false);
                acc[mt][nt] = __builtin_amdgcn_wmma_f32_16x16x32_bf16(
                    false, alo.v, false, bfr[nt].v, (short)0, acc[mt][nt], false, false);
            }
        }
        __syncthreads();
    }

    // C/D 16x16 f32 layout: VGPR j -> row (j + 8*(lane/16)), col = lane%16.
    // Output is write-once/streamed -> nontemporal stores keep X/W panels in L2.
    const int rowoff = (lane >> 4) << 3;  // 0 or 8
    const int coloff = lane & 15;
#pragma unroll
    for (int mt = 0; mt < 4; ++mt) {
#pragma unroll
        for (int nt = 0; nt < 2; ++nt) {
#pragma unroll
            for (int j = 0; j < 8; ++j) {
                const int r = bm + wm + mt * 16 + j + rowoff;
                const int c = bn + wn + nt * 16 + coloff;
                __builtin_nontemporal_store(acc[mt][nt][j], &C[(size_t)r * N + c]);
            }
        }
    }
}

extern "C" void kernel_launch(void* const* d_in, const int* in_sizes, int n_in,
                              void* d_out, int out_size, void* d_ws, size_t ws_size,
                              hipStream_t stream) {
    const float*  X = (const float*)d_in[0];
    const int8_t* W = (const int8_t*)d_in[1];   // int8 weight [K, N]
    float*        C = (float*)d_out;

    const int K = 4096;                 // IN  (reference)
    const int N = 4096;                 // OUT (reference)
    const int M = in_sizes[0] / K;      // B*S = 8192

    dim3 grid(N / BN, M / BM);          // 32 x 64 = 2048 workgroups
    dim3 block(NTHREADS);               // 8 waves (wave32)
    qlinear_wmma_bf16x2<<<grid, block, 0, stream>>>(X, W, C, M, N, K);
}
